// SimpleMultiheadAttention_16028817949504
// MI455X (gfx1250) — compile-verified, matrix operand order checked by reading, not run on hardware
//
#include <hip/hip_runtime.h>
#include <hip/hip_bf16.h>

// ---------------------------------------------------------------------------
// SimpleMultiheadAttention for MI455X (gfx1250), bf16 WMMA, fused 2-pass
// global softmax, async-to-LDS double-buffered tile staging.
// B=4 T=1024 E=768 H=12 D=64.
// ---------------------------------------------------------------------------

typedef __attribute__((ext_vector_type(16))) __bf16 v16bf;
typedef __attribute__((ext_vector_type(8)))  float  v8f;

__device__ __forceinline__ unsigned short f2bf(float f) {
    unsigned int u = __float_as_uint(f);
    unsigned int r = u + 0x7FFFu + ((u >> 16) & 1u);   // round-to-nearest-even
    return (unsigned short)(r >> 16);
}
__device__ __forceinline__ unsigned int f2bf2(float lo, float hi) {
    return (unsigned int)f2bf(lo) | ((unsigned int)f2bf(hi) << 16);
}

__device__ __forceinline__ v8f wmma_bf16(v16bf a, v16bf b, v8f c) {
    return __builtin_amdgcn_wmma_f32_16x16x32_bf16(
        false, a, false, b, (short)0, c, false, false);
}

// Load a 16x32 bf16 fragment (A: [m][k] or B stored column-K-major: [n][k]).
// Matches ISA 16-bit A layout (CDNA5 ISA 7.12.2). rs in ushorts.
__device__ __forceinline__ v16bf ld_frag(const unsigned short* tile, int rs, int kbase) {
    int l   = threadIdx.x & 31;
    int row = l & 15;
    int g   = (l >> 4) * 8;
    const unsigned short* p = tile + (size_t)row * rs + kbase + g;
    union { v16bf v; unsigned int w[8]; } f;
#pragma unroll
    for (int j = 0; j < 4; ++j) {
        f.w[j]     = *(const unsigned int*)(p + 2 * j);        // K = g+2j, g+2j+1
        f.w[4 + j] = *(const unsigned int*)(p + 16 + 2 * j);   // K = 16+g+2j, +1
    }
    return f.v;
}

// CDNA5 async memory->LDS copy, 16 bytes per lane. Tracked by ASYNCcnt (in-order).
__device__ __forceinline__ void async_copy_b128(void* lds_ptr, const void* gptr) {
    unsigned int lds = (unsigned int)(size_t)lds_ptr;   // flat LDS: offset in [31:0]
    asm volatile("global_load_async_to_lds_b128 %0, %1, off"
                 :: "v"(lds), "v"(gptr) : "memory");
}
template <int N>
__device__ __forceinline__ void async_wait() {
    asm volatile("s_wait_asynccnt %0" :: "i"(N) : "memory");
}

// ---------------------------------------------------------------------------
// GEMM: Y[n,o] = sum_e X[n,e] * W[o,e] + bias[o]   (X: [4096,768], W: [768,768])
// XBF16: input dtype. MODE 0: bf16 scatter [B][H][T][D]; MODE 2: bf16 scatter
// [B][H][D][T] (pre-transposed V); MODE 1: f32 [N,768].
// Software-pipelined: next k-step's global loads issue before the WMMAs.
// ---------------------------------------------------------------------------
template <int XBF16, int MODE>
__global__ __launch_bounds__(256)
void gemm_proj(const float* __restrict__ Xf, const unsigned short* __restrict__ Xb,
               const float* __restrict__ W, const float* __restrict__ bias,
               unsigned short* __restrict__ out_scatter, float* __restrict__ out_f32) {
    __shared__ unsigned short As[128][40];   // 80B rows: 16B-aligned, conflict-free
    __shared__ unsigned short Bs[64][40];
    const int t = threadIdx.x;
    const int w = t >> 5;
    const int wm = w & 3, wn = w >> 2;       // 4x2 waves, each 32x32
    const int rowBase = blockIdx.y * 128;
    const int colBase = blockIdx.x * 64;
    v8f acc[2][2] = {};

    float4 ax[4]; uint4 bx[2]; float4 wx[2];
    auto loadG = [&](int k0) {
        if (XBF16) {
#pragma unroll
            for (int i = 0; i < 2; ++i) {            // 512 x 16B chunks
                int idx4 = t + 256 * i;
                int r = idx4 >> 2, c4 = idx4 & 3;
                bx[i] = *(const uint4*)&Xb[(size_t)(rowBase + r) * 768 + k0 + c4 * 8];
            }
        } else {
#pragma unroll
            for (int i = 0; i < 4; ++i) {            // 1024 float4 chunks
                int idx4 = t + 256 * i;
                int r = idx4 >> 3, c = (idx4 & 7) * 4;
                ax[i] = *(const float4*)&Xf[(size_t)(rowBase + r) * 768 + k0 + c];
            }
        }
#pragma unroll
        for (int i = 0; i < 2; ++i) {                // W: 512 float4 chunks
            int idx4 = t + 256 * i;
            int n = idx4 >> 3, c = (idx4 & 7) * 4;
            wx[i] = *(const float4*)&W[(size_t)(colBase + n) * 768 + k0 + c];
        }
    };
    auto storeL = [&]() {
        if (XBF16) {
#pragma unroll
            for (int i = 0; i < 2; ++i) {
                int idx4 = t + 256 * i;
                int r = idx4 >> 2, c4 = idx4 & 3;
                *(uint4*)&As[r][c4 * 8] = bx[i];
            }
        } else {
#pragma unroll
            for (int i = 0; i < 4; ++i) {
                int idx4 = t + 256 * i;
                int r = idx4 >> 3, c = (idx4 & 7) * 4;
                unsigned int* d = (unsigned int*)&As[r][c];
                d[0] = f2bf2(ax[i].x, ax[i].y);
                d[1] = f2bf2(ax[i].z, ax[i].w);
            }
        }
#pragma unroll
        for (int i = 0; i < 2; ++i) {
            int idx4 = t + 256 * i;
            int n = idx4 >> 3, c = (idx4 & 7) * 4;
            unsigned int* d = (unsigned int*)&Bs[n][c];
            d[0] = f2bf2(wx[i].x, wx[i].y);
            d[1] = f2bf2(wx[i].z, wx[i].w);
        }
    };

    loadG(0);
    for (int k0 = 0; k0 < 768; k0 += 32) {
        __syncthreads();                 // previous iteration's frag reads done
        storeL();
        __syncthreads();
        loadG(k0 + 32 < 768 ? k0 + 32 : 0);   // overlap with WMMAs below
        v16bf a0 = ld_frag(&As[wm * 32][0],      40, 0);
        v16bf a1 = ld_frag(&As[wm * 32 + 16][0], 40, 0);
        v16bf b0 = ld_frag(&Bs[wn * 32][0],      40, 0);
        v16bf b1 = ld_frag(&Bs[wn * 32 + 16][0], 40, 0);
        acc[0][0] = wmma_bf16(a0, b0, acc[0][0]);
        acc[0][1] = wmma_bf16(a0, b1, acc[0][1]);
        acc[1][0] = wmma_bf16(a1, b0, acc[1][0]);
        acc[1][1] = wmma_bf16(a1, b1, acc[1][1]);
    }

    int l = t & 31;
    int N = l & 15, g = l >> 4;
#pragma unroll
    for (int i = 0; i < 2; ++i)
#pragma unroll
        for (int j = 0; j < 2; ++j)
#pragma unroll
            for (int r = 0; r < 8; ++r) {
                int gr = rowBase + wm * 32 + 16 * i + r + 8 * g;   // row index
                int o  = colBase + wn * 32 + 16 * j + N;           // out feature
                float v = acc[i][j][r] + bias[o];
                if (MODE == 0) {
                    int b = gr >> 10, tt = gr & 1023;
                    int h = o >> 6,  d  = o & 63;
                    out_scatter[(((size_t)(b * 12 + h) * 1024 + tt) << 6) + d] = f2bf(v);
                } else if (MODE == 2) {                             // V transposed
                    int b = gr >> 10, tt = gr & 1023;
                    int h = o >> 6,  d  = o & 63;
                    out_scatter[(((size_t)(b * 12 + h) * 64 + d) << 10) + tt] = f2bf(v);
                } else {
                    out_f32[(size_t)gr * 768 + o] = v;
                }
            }
}

// ---------------------------------------------------------------------------
// Pass A: per (q-tensor, b, h, 64-row t-chunk) partial (max, sumexp) of
// scores s = (q.k)/8.  Double-buffered async K tiles; tile-max softmax.
// ---------------------------------------------------------------------------
__global__ __launch_bounds__(128)
void attn_stats(const unsigned short* __restrict__ Qb,
                const unsigned short* __restrict__ Kb,
                float* __restrict__ partials) {
    __shared__ unsigned short Ks[2][16][72];   // 144B rows, 16B-aligned chunks
    __shared__ float redM[4], redZ[4];
    int bx = blockIdx.x;
    int chunk = bx & 15;
    int bh = (bx >> 4) % 48;
    int qi = bx / (16 * 48);
    int t = threadIdx.x, w = t >> 5, l = t & 31;

    const unsigned short* Qtile =
        Qb + ((size_t)qi * 48 + bh) * 65536 + (size_t)(chunk * 64 + w * 16) * 64;
    v16bf qa0 = ld_frag(Qtile, 64, 0);
    v16bf qa1 = ld_frag(Qtile, 64, 32);
    const unsigned short* Kbh = Kb + (size_t)bh * 65536;

    const int sK = t >> 3, cK = (t & 7) * 8;   // one 16B chunk per thread
    async_copy_b128(&Ks[0][sK][cK], Kbh + (size_t)sK * 64 + cK);   // tile 0

    float m = -1e30f, Z = 0.f;
    for (int it = 0; it < 64; ++it) {
        int cur = it & 1, nxt = cur ^ 1;
        int s0n = (it < 63 ? it + 1 : it) * 16;           // clamped prefetch
        __syncthreads();                                  // buf[nxt] reads done
        async_copy_b128(&Ks[nxt][sK][cK], Kbh + (size_t)(s0n + sK) * 64 + cK);
        async_wait<1>();                                  // own cur-tile op done
        __syncthreads();                                  // everyone's cur done
        v16bf kb0 = ld_frag(&Ks[cur][0][0], 72, 0);
        v16bf kb1 = ld_frag(&Ks[cur][0][0], 72, 32);
        v8f zc = {};
        v8f sc = wmma_bf16(qa0, kb0, zc);
        sc = wmma_bf16(qa1, kb1, sc);
        float sv[8];
#pragma unroll
        for (int r = 0; r < 8; ++r) sv[r] = sc[r] * 0.125f;     // 1/sqrt(64)
        float tmax = fmaxf(fmaxf(fmaxf(sv[0], sv[1]), fmaxf(sv[2], sv[3])),
                           fmaxf(fmaxf(sv[4], sv[5]), fmaxf(sv[6], sv[7])));
        float nm = fmaxf(m, tmax);
        float es = 0.f;
#pragma unroll
        for (int r = 0; r < 8; ++r) es += __expf(sv[r] - nm);   // independent exps
        Z = Z * __expf(m - nm) + es;
        m = nm;
    }
#pragma unroll
    for (int off = 16; off > 0; off >>= 1) {          // wave32 reduce
        float mo = __shfl_xor(m, off, 32);
        float Zo = __shfl_xor(Z, off, 32);
        float mn = fmaxf(m, mo);
        Z = Z * __expf(m - mn) + Zo * __expf(mo - mn);
        m = mn;
    }
    if (l == 0) { redM[w] = m; redZ[w] = Z; }
    __syncthreads();
    if (t == 0) {
        float M = redM[0], S = redZ[0];
        for (int i = 1; i < 4; ++i) {
            float mn = fmaxf(M, redM[i]);
            S = S * __expf(M - mn) + redZ[i] * __expf(redM[i] - mn);
            M = mn;
        }
        int idx = (qi * 48 + bh) * 16 + chunk;
        partials[2 * idx] = M;
        partials[2 * idx + 1] = S;
    }
}

// Merge chunk partials -> (m_i, c_i = T/(3*Z_i)) per (qi,b,h). 144 entries.
__global__ void attn_reduce(const float* __restrict__ partials, float* __restrict__ stats) {
    int idx = blockIdx.x * blockDim.x + threadIdx.x;
    if (idx >= 144) return;
    float M = -1e30f, S = 0.f;
    for (int c = 0; c < 16; ++c) {
        float m = partials[2 * (idx * 16 + c)];
        float z = partials[2 * (idx * 16 + c) + 1];
        float mn = fmaxf(M, m);
        S = S * __expf(M - mn) + z * __expf(m - mn);
        M = mn;
    }
    stats[2 * idx]     = M;
    stats[2 * idx + 1] = 1024.0f / (3.0f * S);
}

// ---------------------------------------------------------------------------
// Pass B: out[b,h,t,:] = sum_s [sum_i c_i exp(s_i - m_i)] * V[s,:]
// K and (pre-transposed) V tiles both async + double-buffered.
// ---------------------------------------------------------------------------
__global__ __launch_bounds__(128)
void attn_av(const unsigned short* __restrict__ Qb,
             const unsigned short* __restrict__ Kb,
             const unsigned short* __restrict__ Vt,
             const float* __restrict__ stats,
             unsigned short* __restrict__ AO) {
    __shared__ unsigned short Ks[2][32][72];     // [s][d]
    __shared__ unsigned short Vst[2][64][40];    // [d][s]  (V already transposed)
    __shared__ unsigned short Pl[4][16][34];     // per-wave P tile [t][s]
    int bx = blockIdx.x;
    int tchunk = bx & 15;
    int bh = bx >> 4;
    int b = bh / 12, h = bh % 12;
    int t = threadIdx.x, w = t >> 5, l = t & 31;
    int trow = tchunk * 64 + w * 16;

    const unsigned short* Kbh = Kb + (size_t)bh * 65536;
    const unsigned short* Vbh = Vt + (size_t)bh * 65536;

    v16bf qa[3][2];
#pragma unroll
    for (int qi = 0; qi < 3; ++qi) {
        const unsigned short* Qtile =
            Qb + ((size_t)qi * 48 + bh) * 65536 + (size_t)trow * 64;
        qa[qi][0] = ld_frag(Qtile, 64, 0);
        qa[qi][1] = ld_frag(Qtile, 64, 32);
    }
    float mm[3], cc[3];
#pragma unroll
    for (int qi = 0; qi < 3; ++qi) {
        mm[qi] = stats[2 * (qi * 48 + bh)];
        cc[qi] = stats[2 * (qi * 48 + bh) + 1];
    }
    v8f acc[4] = {};

    auto stage = [&](int buf, int s0) {
#pragma unroll
        for (int i = 0; i < 2; ++i) {            // K: 256 x 16B chunks
            int idx = t + 128 * i;
            int s = idx >> 3, c8 = (idx & 7) * 8;
            async_copy_b128(&Ks[buf][s][c8], Kbh + (size_t)(s0 + s) * 64 + c8);
        }
#pragma unroll
        for (int i = 0; i < 2; ++i) {            // V^T: 256 x 16B chunks
            int idx = t + 128 * i;
            int d = idx >> 2, c8 = (idx & 3) * 8;
            async_copy_b128(&Vst[buf][d][c8], Vbh + (size_t)d * 1024 + s0 + c8);
        }
    };

    stage(0, 0);
    for (int it = 0; it < 32; ++it) {
        int cur = it & 1, nxt = cur ^ 1;
        int s0n = (it < 31 ? it + 1 : it) * 32;           // clamped prefetch
        __syncthreads();                                  // buf[nxt] reads done
        stage(nxt, s0n);
        async_wait<4>();                                  // own cur-tile ops done
        __syncthreads();                                  // everyone's cur done
#pragma unroll
        for (int sh = 0; sh < 2; ++sh) {                  // two 16-col s halves
            v16bf kb0 = ld_frag(&Ks[cur][sh * 16][0], 72, 0);
            v16bf kb1 = ld_frag(&Ks[cur][sh * 16][0], 72, 32);
            v8f sc[3];
#pragma unroll
            for (int qi = 0; qi < 3; ++qi) {
                v8f z = {};
                z = wmma_bf16(qa[qi][0], kb0, z);
                z = wmma_bf16(qa[qi][1], kb1, z);
                sc[qi] = z;
            }
#pragma unroll
            for (int r = 0; r < 8; ++r) {
                float p = cc[0] * __expf(sc[0][r] * 0.125f - mm[0])
                        + cc[1] * __expf(sc[1][r] * 0.125f - mm[1])
                        + cc[2] * __expf(sc[2][r] * 0.125f - mm[2]);
                int M = r + 8 * (l >> 4), N = l & 15;
                Pl[w][M][sh * 16 + N] = f2bf(p);
            }
        }
        asm volatile("s_wait_dscnt 0" ::: "memory");      // own-wave P st -> ld
        v16bf pa = ld_frag(&Pl[w][0][0], 34, 0);
#pragma unroll
        for (int jt = 0; jt < 4; ++jt) {
            v16bf vb = ld_frag(&Vst[cur][jt * 16][0], 40, 0);
            acc[jt] = wmma_bf16(pa, vb, acc[jt]);
        }
    }

    int N = l & 15, g = l >> 4;
#pragma unroll
    for (int jt = 0; jt < 4; ++jt)
#pragma unroll
        for (int r = 0; r < 8; ++r) {
            int tr = trow + r + 8 * g;
            int e = h * 64 + jt * 16 + N;
            AO[(size_t)(b * 1024 + tr) * 768 + e] = f2bf(acc[jt][r]);
        }
}

// ---------------------------------------------------------------------------
extern "C" void kernel_launch(void* const* d_in, const int* in_sizes, int n_in,
                              void* d_out, int out_size, void* d_ws, size_t ws_size,
                              hipStream_t stream) {
    (void)in_sizes; (void)n_in; (void)out_size; (void)ws_size;
    const float* q1  = (const float*)d_in[0];
    const float* q2  = (const float*)d_in[1];
    const float* q3  = (const float*)d_in[2];
    const float* key = (const float*)d_in[3];
    const float* val = (const float*)d_in[4];
    const float* Wq1 = (const float*)d_in[5];  const float* bq1 = (const float*)d_in[6];
    const float* Wq2 = (const float*)d_in[7];  const float* bq2 = (const float*)d_in[8];
    const float* Wq3 = (const float*)d_in[9];  const float* bq3 = (const float*)d_in[10];
    const float* Wk  = (const float*)d_in[11]; const float* bk  = (const float*)d_in[12];
    const float* Wv  = (const float*)d_in[13]; const float* bv  = (const float*)d_in[14];
    const float* Wo  = (const float*)d_in[15]; const float* bo  = (const float*)d_in[16];

    const size_t BHTD = 3145728;               // 4*12*1024*64 == 4*1024*768
    unsigned short* Qb = (unsigned short*)d_ws;
    unsigned short* Kb = Qb + 3 * BHTD;
    unsigned short* Vb = Kb + BHTD;            // [B][H][D][T] (transposed)
    unsigned short* AO = Vb + BHTD;
    float* partials = (float*)(AO + BHTD);     // 2304 * 2 floats
    float* stats    = partials + 2 * 2304;     // 144 * 2 floats

    dim3 gg(12, 32), gb(256);
    gemm_proj<0,0><<<gg, gb, 0, stream>>>(q1,  nullptr, Wq1, bq1, Qb,            nullptr);
    gemm_proj<0,0><<<gg, gb, 0, stream>>>(q2,  nullptr, Wq2, bq2, Qb + BHTD,     nullptr);
    gemm_proj<0,0><<<gg, gb, 0, stream>>>(q3,  nullptr, Wq3, bq3, Qb + 2 * BHTD, nullptr);
    gemm_proj<0,0><<<gg, gb, 0, stream>>>(key, nullptr, Wk,  bk,  Kb,            nullptr);
    gemm_proj<0,2><<<gg, gb, 0, stream>>>(val, nullptr, Wv,  bv,  Vb,            nullptr);
    attn_stats<<<2304, 128, 0, stream>>>(Qb, Kb, partials);
    attn_reduce<<<3, 64, 0, stream>>>(partials, stats);
    attn_av<<<768, 128, 0, stream>>>(Qb, Kb, Vb, stats, AO);
    gemm_proj<1,1><<<gg, gb, 0, stream>>>(nullptr, AO, Wo, bo, nullptr, (float*)d_out);
}
